// FlashAttention_65773129171310
// MI455X (gfx1250) — compile-verified, matrix operand order checked by reading, not run on hardware
//
#include <hip/hip_runtime.h>
#include <math.h>

// ---- problem constants (match the JAX reference) ----
#define DIMX   1024
#define NHEAD  16
#define HDIM   64
#define SEQ    2048
#define BATCH  2
#define MTOK   (BATCH * SEQ)          // 4096 tokens
#define SCALE  0.125f                 // 64^-0.5

typedef __attribute__((ext_vector_type(16))) _Float16 v16h;
typedef __attribute__((ext_vector_type(8)))  _Float16 v8h;
typedef __attribute__((ext_vector_type(8)))  float    v8f;

union V8F  { v8f  v; float    f[8];  };
union H8   { v8h  v; _Float16 e[8];  };
union V16H { v16h v; v8h h2[2]; _Float16 e[16]; };

// ---------------------------------------------------------------------------
// CDNA5 async copy: LDS[vdst_lane_byte_off] <= MEM[gaddr], 16 bytes, tracked
// by ASYNCcnt (ISA 10.?: GLOBAL_LOAD_ASYNC_TO_LDS_B128, GV mode).
// Generic pointers to __shared__ have the LDS byte offset in their low 32 bits.
// ---------------------------------------------------------------------------
__device__ __forceinline__ void async_copy_b128(void* lds_dst, const void* gsrc) {
  unsigned int loff = (unsigned int)(unsigned long long)lds_dst;
  unsigned long long ga = (unsigned long long)gsrc;
  asm volatile("global_load_async_to_lds_b128 %0, %1, off"
               :: "v"(loff), "v"(ga) : "memory");
}
__device__ __forceinline__ void wait_async0() {
  asm volatile("s_wait_asynccnt 0x0" ::: "memory");
}

// ---------------------------------------------------------------------------
// 16x32 f16 fragment load (A-operand layout; B column-major uses the same
// pattern).  Lane r (r = lane&15) supplies row r; kbase = (lane>>4)*8:
//   halves 0..7  <- K = kbase .. kbase+7
//   halves 8..15 <- K = kbase+16 .. kbase+23
// Works for global or LDS base (addrspace inferred after inlining -> ds_load).
// ---------------------------------------------------------------------------
__device__ __forceinline__ v16h load_frag16x32(const _Float16* __restrict__ base,
                                               int ld) {
  const int lane = threadIdx.x & 31;
  const int row  = lane & 15;
  const int kb   = (lane >> 4) << 3;
  const _Float16* p = base + (size_t)row * ld + kb;
  V16H u;
  u.h2[0] = *(const v8h*)(p);
  u.h2[1] = *(const v8h*)(p + 16);
  return u.v;
}

__device__ __forceinline__ v8f wmma_f16(v16h a, v16h b, v8f c) {
  return __builtin_amdgcn_wmma_f32_16x16x32_f16(
      false, a, false, b, (short)0, c, false, false);
}

// ---------------------------------------------------------------------------
// fp32 -> f16 conversion
// ---------------------------------------------------------------------------
__global__ void cvt_f32_to_f16(const float* __restrict__ in,
                               _Float16* __restrict__ out, int n) {
  int i = blockIdx.x * blockDim.x + threadIdx.x;
  if (i < n) out[i] = (_Float16)in[i];
}

// ---------------------------------------------------------------------------
// C[M,N] = A[M,K] * Bw[N,K]^T + bias[N]    (torch Linear convention)
// Block: 256 threads = 8 waves.  Block tile 256x64, wave tile 32x64.
// K-steps of 32 staged through double-buffered LDS with async copies:
//   lA: 256x32 f16 (16 KB)  lB: 64x32 f16 (4 KB)   x2 buffers = 40 KB.
// mode 0: f16 out row-major | 1: f16 V head-transposed | 2: f32 out row-major
// ---------------------------------------------------------------------------
#define GBM 256
#define GBN 64
#define GBK 32

__global__ __launch_bounds__(256)
void gemm_bias_wmma(const _Float16* __restrict__ A,
                    const _Float16* __restrict__ Bw,
                    const float* __restrict__ bias,
                    void* __restrict__ out,
                    int M, int N, int K, int mode) {
  __shared__ _Float16 lA[2][GBM * GBK];
  __shared__ _Float16 lB[2][GBN * GBK];

  const int tid  = threadIdx.x;
  const int wave = tid >> 5;
  const int lane = tid & 31;
  const int n0 = blockIdx.x * GBN;
  const int mb = blockIdx.y * GBM;

  // cooperative tile copy: A = 1024 b128 chunks (4/thread), B = 256 (1/thread)
  auto copy_tile = [&](int buf, int k0) {
#pragma unroll
    for (int i = 0; i < 4; ++i) {
      const int c   = tid + i * 256;          // 0..1023
      const int row = c >> 2, ch = (c & 3) << 3;
      async_copy_b128(&lA[buf][row * GBK + ch],
                      A + (size_t)(mb + row) * K + k0 + ch);
    }
    {
      const int row = tid >> 2, ch = (tid & 3) << 3;
      async_copy_b128(&lB[buf][row * GBK + ch],
                      Bw + (size_t)(n0 + row) * K + k0 + ch);
    }
  };

  v8f acc[2][4] = {};

  copy_tile(0, 0);
  wait_async0();
  __syncthreads();

  const int KSTEPS = K / GBK;
  for (int kb = 0; kb < KSTEPS; ++kb) {
    const int cur = kb & 1;
    if (kb + 1 < KSTEPS) copy_tile(cur ^ 1, (kb + 1) * GBK);

    const _Float16* Ab = &lA[cur][(wave * 32) * GBK];
    const _Float16* Bb = &lB[cur][0];
    v16h a0 = load_frag16x32(Ab, GBK);
    v16h a1 = load_frag16x32(Ab + 16 * GBK, GBK);
    v16h b0 = load_frag16x32(Bb +  0 * GBK, GBK);
    v16h b1 = load_frag16x32(Bb + 16 * GBK, GBK);
    v16h b2 = load_frag16x32(Bb + 32 * GBK, GBK);
    v16h b3 = load_frag16x32(Bb + 48 * GBK, GBK);
    acc[0][0] = wmma_f16(a0, b0, acc[0][0]);
    acc[0][1] = wmma_f16(a0, b1, acc[0][1]);
    acc[0][2] = wmma_f16(a0, b2, acc[0][2]);
    acc[0][3] = wmma_f16(a0, b3, acc[0][3]);
    acc[1][0] = wmma_f16(a1, b0, acc[1][0]);
    acc[1][1] = wmma_f16(a1, b1, acc[1][1]);
    acc[1][2] = wmma_f16(a1, b2, acc[1][2]);
    acc[1][3] = wmma_f16(a1, b3, acc[1][3]);

    wait_async0();        // next buffer resident (this wave's copies)
    __syncthreads();      // ... and everyone else's; current buffer free
  }

  const int m0  = mb + wave * 32;
  const int col = lane & 15;            // N within 16-tile (C/D layout)
  const int rb  = (lane >> 4) << 3;     // row offset 0 or 8 within 16-tile

#pragma unroll
  for (int j = 0; j < 4; ++j) {
    const float bn = bias[n0 + j * 16 + col];
#pragma unroll
    for (int i = 0; i < 2; ++i) acc[i][j] = acc[i][j] + bn;
  }

  if (mode == 0) {                       // f16 row-major
    _Float16* O = (_Float16*)out;
#pragma unroll
    for (int i = 0; i < 2; ++i)
#pragma unroll
      for (int j = 0; j < 4; ++j) {
        V8F u; u.v = acc[i][j];
        const int m = m0 + i * 16 + rb;
        const int n = n0 + j * 16 + col;
#pragma unroll
        for (int r = 0; r < 8; ++r)
          O[(size_t)(m + r) * N + n] = (_Float16)u.f[r];
      }
  } else if (mode == 1) {                // V: [B*NHEAD, HDIM, SEQ] transposed
    _Float16* O = (_Float16*)out;
#pragma unroll
    for (int i = 0; i < 2; ++i)
#pragma unroll
      for (int j = 0; j < 4; ++j) {
        V8F u; u.v = acc[i][j];
        const int m = m0 + i * 16 + rb;        // token (contiguous out dim)
        const int n = n0 + j * 16 + col;       // feature
        const int b = m >> 11, s = m & (SEQ - 1);
        const int h = n >> 6,  d = n & (HDIM - 1);
        H8 pk;
#pragma unroll
        for (int r = 0; r < 8; ++r) pk.e[r] = (_Float16)u.f[r];
        *(v8h*)(O + ((size_t)((b * NHEAD + h) * HDIM + d)) * SEQ + s) = pk.v;
      }
  } else {                               // f32 row-major (final output)
    float* O = (float*)out;
#pragma unroll
    for (int i = 0; i < 2; ++i)
#pragma unroll
      for (int j = 0; j < 4; ++j) {
        V8F u; u.v = acc[i][j];
        const int m = m0 + i * 16 + rb;
        const int n = n0 + j * 16 + col;
#pragma unroll
        for (int r = 0; r < 8; ++r)
          O[(size_t)(m + r) * N + n] = u.f[r];
      }
  }
}

// ---------------------------------------------------------------------------
// Flash attention.  Block = 8 waves = 128 queries of one (batch,head); all
// waves share K/V, so each 32-key block is staged once into double-buffered
// LDS via async copies (4 KB K + 4 KB V per buffer).
// Scores computed transposed (S^T = K_blk * Q^T) so softmax stats are
// per-lane (+ one shfl_xor(16)), and P^T in C-layout is directly the
// B-operand fragment for O^T = V^T * P^T (half r of fragment = C VGPR r).
// ---------------------------------------------------------------------------
__global__ __launch_bounds__(256)
void attn_wmma(const _Float16* __restrict__ Q,   // [B,SEQ,DIMX] f16
               const _Float16* __restrict__ Kc,  // [B,SEQ,DIMX] f16
               const _Float16* __restrict__ Vt,  // [B*NHEAD, HDIM, SEQ] f16
               _Float16* __restrict__ O) {       // [B,SEQ,DIMX] f16
  __shared__ _Float16 lK[2][32 * HDIM];          // rows = key, ld = 64
  __shared__ _Float16 lV[2][HDIM * 32];          // rows = d,   ld = 32

  const int tid  = threadIdx.x;
  const int wave = tid >> 5;
  const int lane = tid & 31;
  const int qt = blockIdx.x & 15;        // 16 query tiles of 128
  const int bh = blockIdx.x >> 4;        // 0..31
  const int b  = bh >> 4;
  const int h  = bh & 15;
  const int q0 = qt * 128 + wave * 16;

  const _Float16* Kg = Kc + ((size_t)b * SEQ) * DIMX + h * HDIM;
  const _Float16* Vg = Vt + (size_t)bh * HDIM * SEQ;

  // 256 b128 chunks each for K and V -> 1 chunk per thread per tile
  auto copy_tile = [&](int buf, int s0) {
    {
      const int row = tid >> 3, ch = (tid & 7) << 3;     // K: 32 x 64
      async_copy_b128(&lK[buf][row * HDIM + ch],
                      Kg + (size_t)(s0 + row) * DIMX + ch);
    }
    {
      const int row = tid >> 2, ch = (tid & 3) << 3;     // V^T: 64 x 32
      async_copy_b128(&lV[buf][row * 32 + ch],
                      Vg + (size_t)row * SEQ + s0 + ch);
    }
  };

  // Q fragments (B operand: lane = query row, contraction over head dim)
  const _Float16* Qb = Q + ((size_t)(b * SEQ + q0)) * DIMX + h * HDIM;
  const v16h qf0 = load_frag16x32(Qb, DIMX);
  const v16h qf1 = load_frag16x32(Qb + 32, DIMX);

  v8f   o[4] = {};                       // O^T tiles: d = t*16.., lane = q
  float m_run = -1e30f, l_run = 0.f;

  copy_tile(0, 0);
  wait_async0();
  __syncthreads();

  for (int sb = 0; sb < SEQ / 32; ++sb) {
    const int cur = sb & 1;
    if (sb + 1 < SEQ / 32) copy_tile(cur ^ 1, (sb + 1) * 32);

    // --- S^T(32 keys x 16 queries) = K_blk * Q^T, contraction over HDIM ---
    v16h kf0 = load_frag16x32(&lK[cur][0], HDIM);
    v16h kf1 = load_frag16x32(&lK[cur][32], HDIM);
    v16h kf2 = load_frag16x32(&lK[cur][16 * HDIM], HDIM);
    v16h kf3 = load_frag16x32(&lK[cur][16 * HDIM + 32], HDIM);
    v8f st0 = {}, st1 = {};
    st0 = wmma_f16(kf0, qf0, st0);  st0 = wmma_f16(kf1, qf1, st0);
    st1 = wmma_f16(kf2, qf0, st1);  st1 = wmma_f16(kf3, qf1, st1);

    // --- online softmax along keys (per lane + one xor16 across halves) ---
    V8F u0, u1; u0.v = st0; u1.v = st1;
    float mloc = -1e30f;
#pragma unroll
    for (int r = 0; r < 8; ++r) {
      u0.f[r] *= SCALE; u1.f[r] *= SCALE;
      mloc = fmaxf(mloc, fmaxf(u0.f[r], u1.f[r]));
    }
    mloc = fmaxf(mloc, __shfl_xor(mloc, 16, 32));
    const float m_new = fmaxf(m_run, mloc);
    const float alpha = __expf(m_run - m_new);
    float lsum = 0.f;
    V16H pf;
#pragma unroll
    for (int r = 0; r < 8; ++r) {
      const float p0 = __expf(u0.f[r] - m_new);
      const float p1 = __expf(u1.f[r] - m_new);
      lsum += p0 + p1;
      pf.e[r]     = (_Float16)p0;        // keys s0 + kb..      (tile0 rows)
      pf.e[8 + r] = (_Float16)p1;        // keys s0+16 + kb..   (tile1 rows)
    }
    lsum  = lsum + __shfl_xor(lsum, 16, 32);
    l_run = l_run * alpha + lsum;
    m_run = m_new;

    // --- O^T += V^T * P^T  (A = V^T rows d, contraction = 32 keys) ---
#pragma unroll
    for (int t = 0; t < 4; ++t) {
      o[t] = o[t] * alpha;
      v16h vf = load_frag16x32(&lV[cur][t * 16 * 32], 32);
      o[t] = wmma_f16(vf, pf.v, o[t]);
    }

    wait_async0();
    __syncthreads();
  }

  // --- normalize + store (O^T tile: lane -> q, VGPR -> d, contiguous) ---
  const float inv = 1.0f / l_run;
  const int col = lane & 15;             // query within tile
  const int rb  = (lane >> 4) << 3;      // d offset 0 or 8
  _Float16* Ob = O + ((size_t)(b * SEQ + q0 + col)) * DIMX + h * HDIM;
#pragma unroll
  for (int t = 0; t < 4; ++t) {
    V8F u; u.v = o[t];
    H8 pk;
#pragma unroll
    for (int r = 0; r < 8; ++r) pk.e[r] = (_Float16)(u.f[r] * inv);
    *(v8h*)(Ob + t * 16 + rb) = pk.v;
  }
}

// ---------------------------------------------------------------------------
// host-side launcher
// ---------------------------------------------------------------------------
extern "C" void kernel_launch(void* const* d_in, const int* in_sizes, int n_in,
                              void* d_out, int out_size, void* d_ws, size_t ws_size,
                              hipStream_t stream) {
  const float* x  = (const float*)d_in[0];
  const float* wq = (const float*)d_in[1];
  const float* bq = (const float*)d_in[2];
  const float* wk = (const float*)d_in[3];
  const float* bk = (const float*)d_in[4];
  const float* wv = (const float*)d_in[5];
  const float* bv = (const float*)d_in[6];
  const float* wo = (const float*)d_in[7];
  const float* bo = (const float*)d_in[8];

  const int NX = MTOK * DIMX;            // 4,194,304
  const int NW = DIMX * DIMX;            // 1,048,576

  _Float16* xh  = (_Float16*)d_ws;       // 48 MB of f16 scratch total
  _Float16* wqh = xh  + (size_t)NX;
  _Float16* wkh = wqh + (size_t)NW;
  _Float16* wvh = wkh + (size_t)NW;
  _Float16* woh = wvh + (size_t)NW;
  _Float16* Qh  = woh + (size_t)NW;
  _Float16* Kh  = Qh  + (size_t)NX;
  _Float16* Vt  = Kh  + (size_t)NX;      // [B*NHEAD, HDIM, SEQ]
  _Float16* Ah  = Vt  + (size_t)NX;      // attention output, f16

  cvt_f32_to_f16<<<(NX + 255) / 256, 256, 0, stream>>>(x,  xh,  NX);
  cvt_f32_to_f16<<<(NW + 255) / 256, 256, 0, stream>>>(wq, wqh, NW);
  cvt_f32_to_f16<<<(NW + 255) / 256, 256, 0, stream>>>(wk, wkh, NW);
  cvt_f32_to_f16<<<(NW + 255) / 256, 256, 0, stream>>>(wv, wvh, NW);
  cvt_f32_to_f16<<<(NW + 255) / 256, 256, 0, stream>>>(wo, woh, NW);

  dim3 g(DIMX / GBN, MTOK / GBM);        // (16,16)
  gemm_bias_wmma<<<g, 256, 0, stream>>>(xh, wqh, bq, Qh, MTOK, DIMX, DIMX, 0);
  gemm_bias_wmma<<<g, 256, 0, stream>>>(xh, wkh, bk, Kh, MTOK, DIMX, DIMX, 0);
  gemm_bias_wmma<<<g, 256, 0, stream>>>(xh, wvh, bv, Vt, MTOK, DIMX, DIMX, 1);

  attn_wmma<<<BATCH * NHEAD * (SEQ / 128), 256, 0, stream>>>(Qh, Kh, Vt, Ah);

  gemm_bias_wmma<<<g, 256, 0, stream>>>(Ah, woh, bo, (float*)d_out,
                                        MTOK, DIMX, DIMX, 2);
}
